// CausalSelfAttentionWithKVCache_27779848471171
// MI455X (gfx1250) — matrix operand order repaired and checked by reading, not compiled, not run on hardware
//
#include <hip/hip_runtime.h>
#include <hip/hip_bf16.h>

typedef __bf16 bf16;
typedef __attribute__((ext_vector_type(16))) __bf16 v16bf;
typedef __attribute__((ext_vector_type(8)))  __bf16 v8bf;
typedef __attribute__((ext_vector_type(4)))  __bf16 v4bf;
typedef __attribute__((ext_vector_type(8)))  float  v8f;
typedef __attribute__((ext_vector_type(4)))  unsigned int u32x4;
typedef __attribute__((ext_vector_type(8)))  int i32x8;
typedef __attribute__((ext_vector_type(4)))  int i32x4;

#define WMMA_BF16(a, b, c) \
  __builtin_amdgcn_wmma_f32_16x16x32_bf16(false, (a), false, (b), (short)0, (c), false, false)

#define NEG_INF (-__builtin_inff())

#if __has_builtin(__builtin_amdgcn_tensor_load_to_lds) && __has_builtin(__builtin_amdgcn_s_wait_tensorcnt)
#define USE_TDM 1
#else
#define USE_TDM 0
#endif

static constexpr int Bn = 4;
static constexpr int Tn = 2048;
static constexpr int Cn = 1024;
static constexpr int Hn = 16;
static constexpr int Dn = 64;
static constexpr int Mrows = Bn * Tn;   // 8192
static constexpr int KP = 512;          // k-panel staged in LDS (32 rows x 512 = 32 KB)

static __device__ __forceinline__ v8bf ld8(const bf16* p) {
  return *(const v8bf*)p;               // global_load_b128 / ds_load_b128
}
static __device__ __forceinline__ v16bf mk16(v8bf lo, v8bf hi) {
  return __builtin_shufflevector(lo, hi, 0,1,2,3,4,5,6,7,8,9,10,11,12,13,14,15);
}

// ---------------------------------------------------------------------------
// TDM: stage a 2-D tile (rows x tile_k bf16, row stride = stride_k elements)
// from global into LDS (contiguous row-major). Descriptor per ISA 08 §8.
// Caller: wave-uniform args, issued by one wave, followed by block barrier.
// ---------------------------------------------------------------------------
static __device__ __forceinline__ void tdm_stage_rows(
    const bf16* gsrc, bf16* ldst, int tile_k, int rows, int stride_k)
{
#if USE_TDM
  const unsigned long long ga = (unsigned long long)(__SIZE_TYPE__)gsrc;
  u32x4 g0;
  g0[0] = 1u;                                   // count=1 (valid), user mode
  g0[1] = (unsigned)(__SIZE_TYPE__)ldst;        // lds_addr (bytes)
  g0[2] = (unsigned)ga;                         // global_addr[31:0]
  g0[3] = (unsigned)(ga >> 32) | (2u << 30);    // global_addr[56:32] | type=2

  i32x8 g1;
  g1[0] = (int)(1u << 16);                                          // data_size=2B
  g1[1] = (int)(((unsigned)stride_k & 0xFFFFu) << 16);              // tensor_dim0[15:0]
  g1[2] = (int)((((unsigned)stride_k >> 16) & 0xFFFFu)
              | (((unsigned)Mrows & 0xFFFFu) << 16));               // td0[31:16] | tensor_dim1[15:0]
  g1[3] = (int)(((((unsigned)Mrows >> 16) & 0xFFFFu))
              | (((unsigned)tile_k & 0xFFFFu) << 16));              // td1[31:16] | tile_dim0
  g1[4] = (int)((unsigned)rows & 0xFFFFu);                          // tile_dim1 (tile_dim2=0)
  g1[5] = (int)(unsigned)stride_k;                                  // tensor_dim0_stride[31:0]
  g1[6] = 0;
  g1[7] = 0;

  const i32x4 z4 = {0, 0, 0, 0};
#if __clang_major__ >= 23
  const i32x8 z8 = {0, 0, 0, 0, 0, 0, 0, 0};
  __builtin_amdgcn_tensor_load_to_lds(g0, g1, z4, z4, z8, 0);
#else
  __builtin_amdgcn_tensor_load_to_lds(g0, g1, z4, z4, 0);
#endif
  __builtin_amdgcn_s_wait_tensorcnt(0);
#else
  // Fallback: synchronous copy by the calling wave.
  const int lane = threadIdx.x & 31;
  const int total = tile_k * rows;
  for (int i = lane * 8; i < total; i += 32 * 8) {
    const int r = i / tile_k;
    const int c = i - r * tile_k;
    *(v8bf*)(ldst + i) = ld8(gsrc + (__SIZE_TYPE__)r * stride_k + c);
  }
#endif
}

// ---------------------------------------------------------------------------
// Kernel 0a: x fp32 -> bf16.
// ---------------------------------------------------------------------------
__global__ __launch_bounds__(256) void xconv_kernel(
    const float* __restrict__ x, bf16* __restrict__ xb)
{
  const size_t i = ((size_t)blockIdx.x * 256 + threadIdx.x) * 4;
  const float4 f = *(const float4*)(x + i);
  v4bf o;
  o[0] = (bf16)f.x; o[1] = (bf16)f.y; o[2] = (bf16)f.z; o[3] = (bf16)f.w;
  *(v4bf*)(xb + i) = o;
}

// ---------------------------------------------------------------------------
// Kernel 0b: W[k][n] fp32 -> Wt[z][n][k] bf16 (LDS-tiled transpose).
// ---------------------------------------------------------------------------
__global__ __launch_bounds__(256) void wconv_kernel(
    const float* __restrict__ Wq, const float* __restrict__ Wk,
    const float* __restrict__ Wv, const float* __restrict__ Wp,
    bf16* __restrict__ Wt)
{
  __shared__ float tile[32][33];
  const int tx = threadIdx.x;   // 0..31
  const int ty = threadIdx.y;   // 0..7
  const int z  = blockIdx.z;
  const float* W = (z == 0) ? Wq : (z == 1) ? Wk : (z == 2) ? Wv : Wp;
  const int kt = blockIdx.x * 32;
  const int nt = blockIdx.y * 32;

  #pragma unroll
  for (int i = 0; i < 4; ++i)
    tile[ty + 8 * i][tx] = W[(size_t)(kt + ty + 8 * i) * Cn + nt + tx];
  __syncthreads();

  bf16* o = Wt + (size_t)z * Cn * Cn;
  #pragma unroll
  for (int i = 0; i < 4; ++i)
    o[(size_t)(nt + ty + 8 * i) * Cn + kt + tx] = (bf16)tile[tx][ty + 8 * i];
}

// ---------------------------------------------------------------------------
// Kernel 1: QKV projection. Block = 8 waves sharing one 32-row A strip which
// is TDM-staged into LDS in two 512-wide k-panels; each wave owns a 32x32
// output block (2x2 WMMA tiles). B streams from global as b128 clauses.
// Outputs: q,k -> [B,H,T,d]; v -> [B,H,d,T].
// ---------------------------------------------------------------------------
__global__ __launch_bounds__(256) void qkv_gemm_kernel(
    const bf16* __restrict__ xb, const bf16* __restrict__ Wt,
    const float* __restrict__ bq, const float* __restrict__ bk,
    const float* __restrict__ bv,
    bf16* __restrict__ qo, bf16* __restrict__ ko, bf16* __restrict__ vo)
{
  __shared__ __align__(16) bf16 tA[32 * KP];   // 32 KB

  const int lane = threadIdx.x & 31;
  const int wave = threadIdx.x >> 5;
  const int hi   = lane >> 4;
  const int ml   = lane & 15;

  const int which = blockIdx.y;                     // 0=q 1=k 2=v
  const int tm    = blockIdx.x >> 2;                // 256 M-tiles of 32 rows
  const int tn    = (blockIdx.x & 3) * 8 + wave;    // 32 N-tiles of 32 cols

  const bf16*  W    = Wt + (size_t)which * Cn * Cn;
  const float* bias = (which == 0) ? bq : (which == 1) ? bk : bv;

  const int colB0 = tn * 32 + ml;
  const int colB1 = colB0 + 16;

  v8f c00, c01, c10, c11;
  #pragma unroll
  for (int i = 0; i < 8; ++i) { c00[i] = 0.f; c01[i] = 0.f; c10[i] = 0.f; c11[i] = 0.f; }

  const bf16* pb0 = W + (size_t)colB0 * Cn + hi * 16;
  const bf16* pb1 = W + (size_t)colB1 * Cn + hi * 16;

  for (int kp = 0; kp < Cn; kp += KP) {
    __syncthreads();   // previous panel fully consumed
    if (wave == 0)
      tdm_stage_rows(xb + (size_t)(tm * 32) * Cn + kp, tA, KP, 32, Cn);
    __syncthreads();   // panel visible to all waves

    const bf16* ta0 = tA + ml * KP + hi * 16;
    const bf16* ta1 = ta0 + 16 * KP;
    #pragma unroll 4
    for (int k0 = 0; k0 < KP; k0 += 32) {
      __builtin_prefetch(pb0 + kp + k0 + 32, 0, 3);
      __builtin_prefetch(pb1 + kp + k0 + 32, 0, 3);
      const v16bf a0 = mk16(ld8(ta0 + k0), ld8(ta0 + k0 + 8));
      const v16bf a1 = mk16(ld8(ta1 + k0), ld8(ta1 + k0 + 8));
      const v16bf b0 = mk16(ld8(pb0 + kp + k0), ld8(pb0 + kp + k0 + 8));
      const v16bf b1 = mk16(ld8(pb1 + kp + k0), ld8(pb1 + kp + k0 + 8));
      c00 = WMMA_BF16(a0, b0, c00);
      c01 = WMMA_BF16(a0, b1, c01);
      c10 = WMMA_BF16(a1, b0, c10);
      c11 = WMMA_BF16(a1, b1, c11);
    }
  }

  bf16* out = (which == 0) ? qo : ko;
  #pragma unroll
  for (int tj = 0; tj < 2; ++tj) {
    const int col = (tj == 0) ? colB0 : colB1;
    const int h   = col >> 6;
    const int dd  = col & 63;
    const float bn = bias[col];
    #pragma unroll
    for (int ti = 0; ti < 2; ++ti) {
      const v8f& acc = (ti == 0) ? ((tj == 0) ? c00 : c01)
                                 : ((tj == 0) ? c10 : c11);
      #pragma unroll
      for (int r = 0; r < 8; ++r) {
        const int row = tm * 32 + ti * 16 + r + 8 * hi;   // b*T + t
        const int bb  = row >> 11;
        const int t   = row & 2047;
        const bf16 val = (bf16)(acc[r] + bn);
        if (which == 2)
          vo[((size_t)(bb * Hn + h) * Dn + dd) * Tn + t] = val;   // [B,H,d,T]
        else
          out[((size_t)(bb * Hn + h) * Tn + t) * Dn + dd] = val;  // [B,H,T,d]
      }
    }
  }
}

// ---------------------------------------------------------------------------
// Kernel 2: flash attention (unchanged structure; WMMA-dense, b128 operands).
// ---------------------------------------------------------------------------
__global__ __launch_bounds__(128) void attn_kernel(
    const bf16* __restrict__ q, const bf16* __restrict__ k,
    const bf16* __restrict__ v, bf16* __restrict__ ao)
{
  __shared__ __align__(16) bf16 plds[4][16 * 32];

  const int lane = threadIdx.x & 31;
  const int wave = threadIdx.x >> 5;
  const int hi   = lane >> 4;
  const int ml   = lane & 15;

  const int bh    = blockIdx.y;   // b*H + h
  const int bb    = bh >> 4;
  const int hh    = bh & 15;
  const int qbase = blockIdx.x * 64 + wave * 16;

  const bf16* Q  = q + (size_t)bh * Tn * Dn;   // [T][d]
  const bf16* K  = k + (size_t)bh * Tn * Dn;   // [T][d]
  const bf16* Vt = v + (size_t)bh * Dn * Tn;   // [d][T]
  bf16* pw = plds[wave];

  const bf16* qr = Q + (size_t)(qbase + ml) * Dn;
  const v16bf aq0 = mk16(ld8(qr + hi * 16),      ld8(qr + 8 + hi * 16));
  const v16bf aq1 = mk16(ld8(qr + 32 + hi * 16), ld8(qr + 40 + hi * 16));

  v8f y0, y1, y2, y3;
  float mrow[8], lrow[8];
  #pragma unroll
  for (int r = 0; r < 8; ++r) {
    y0[r] = 0.f; y1[r] = 0.f; y2[r] = 0.f; y3[r] = 0.f;
    mrow[r] = NEG_INF; lrow[r] = 0.f;
  }

  const int smax = qbase + 16;
  for (int s0 = 0; s0 < smax; s0 += 32) {
    const bf16* kr0 = K + (size_t)(s0 + ml) * Dn;
    const bf16* kr1 = K + (size_t)(s0 + 16 + ml) * Dn;
    __builtin_prefetch(kr0 + 32 * Dn, 0, 2);
    const v16bf bk00 = mk16(ld8(kr0 + hi * 16),      ld8(kr0 + hi * 16 + 8));
    const v16bf bk01 = mk16(ld8(kr0 + 32 + hi * 16), ld8(kr0 + 40 + hi * 16));
    const v16bf bk10 = mk16(ld8(kr1 + hi * 16),      ld8(kr1 + hi * 16 + 8));
    const v16bf bk11 = mk16(ld8(kr1 + 32 + hi * 16), ld8(kr1 + 40 + hi * 16));

    v8f c0, c1;
    #pragma unroll
    for (int i = 0; i < 8; ++i) { c0[i] = 0.f; c1[i] = 0.f; }
    c0 = WMMA_BF16(aq0, bk00, c0);
    c0 = WMMA_BF16(aq1, bk01, c0);
    c1 = WMMA_BF16(aq0, bk10, c1);
    c1 = WMMA_BF16(aq1, bk11, c1);

    float scl[8];
    #pragma unroll
    for (int r = 0; r < 8; ++r) {
      const int qi  = qbase + r + 8 * hi;
      const float s0v = ((s0 + ml)      <= qi) ? c0[r] * 0.125f : NEG_INF;
      const float s1v = ((s0 + 16 + ml) <= qi) ? c1[r] * 0.125f : NEG_INF;

      float mx = fmaxf(s0v, s1v);
      #pragma unroll
      for (int off = 1; off < 16; off <<= 1)
        mx = fmaxf(mx, __shfl_xor(mx, off, 32));
      const float mnew = fmaxf(mrow[r], mx);

      const float p0 = __expf(s0v - mnew);
      const float p1 = __expf(s1v - mnew);
      float rs = p0 + p1;
      #pragma unroll
      for (int off = 1; off < 16; off <<= 1)
        rs += __shfl_xor(rs, off, 32);

      const float corr = __expf(mrow[r] - mnew);
      lrow[r] = lrow[r] * corr + rs;
      mrow[r] = mnew;
      scl[r]  = corr;

      const int m = r + 8 * hi;
      pw[m * 32 + ml]      = (bf16)p0;
      pw[m * 32 + 16 + ml] = (bf16)p1;
    }

    #pragma unroll
    for (int r = 0; r < 8; ++r) {
      y0[r] *= scl[r]; y1[r] *= scl[r]; y2[r] *= scl[r]; y3[r] *= scl[r];
    }

    const bf16* pr = pw + ml * 32;
    const v16bf ap = mk16(ld8(pr + hi * 16), ld8(pr + 8 + hi * 16));

    const bf16* vr = Vt + (size_t)ml * Tn + s0 + hi * 16;
    const v16bf bv0 = mk16(ld8(vr),           ld8(vr + 8));
    const v16bf bv1 = mk16(ld8(vr + 16 * Tn), ld8(vr + 16 * Tn + 8));
    const v16bf bv2 = mk16(ld8(vr + 32 * Tn), ld8(vr + 32 * Tn + 8));
    const v16bf bv3 = mk16(ld8(vr + 48 * Tn), ld8(vr + 48 * Tn + 8));
    y0 = WMMA_BF16(ap, bv0, y0);
    y1 = WMMA_BF16(ap, bv1, y1);
    y2 = WMMA_BF16(ap, bv2, y2);
    y3 = WMMA_BF16(ap, bv3, y3);
  }

  #pragma unroll
  for (int r = 0; r < 8; ++r) {
    const int qi = qbase + r + 8 * hi;
    const float inv = 1.0f / lrow[r];
    bf16* o = ao + (size_t)(bb * Tn + qi) * Cn + hh * Dn + ml;
    o[0]  = (bf16)(y0[r] * inv);
    o[16] = (bf16)(y1[r] * inv);
    o[32] = (bf16)(y2[r] * inv);
    o[48] = (bf16)(y3[r] * inv);
  }
}

// ---------------------------------------------------------------------------
// Kernel 3: output projection, TDM-staged A strip, fp32 result + bias.
// ---------------------------------------------------------------------------
__global__ __launch_bounds__(256) void proj_gemm_kernel(
    const bf16* __restrict__ ain, const bf16* __restrict__ Wt,
    const float* __restrict__ bp, float* __restrict__ out)
{
  __shared__ __align__(16) bf16 tA[32 * KP];   // 32 KB

  const int lane = threadIdx.x & 31;
  const int wave = threadIdx.x >> 5;
  const int hi   = lane >> 4;
  const int ml   = lane & 15;

  const int tm = blockIdx.x >> 2;
  const int tn = (blockIdx.x & 3) * 8 + wave;

  const bf16* W = Wt + (size_t)3 * Cn * Cn;   // Wp

  const int colB0 = tn * 32 + ml;
  const int colB1 = colB0 + 16;

  v8f c00, c01, c10, c11;
  #pragma unroll
  for (int i = 0; i < 8; ++i) { c00[i] = 0.f; c01[i] = 0.f; c10[i] = 0.f; c11[i] = 0.f; }

  const bf16* pb0 = W + (size_t)colB0 * Cn + hi * 16;
  const bf16* pb1 = W + (size_t)colB1 * Cn + hi * 16;

  for (int kp = 0; kp < Cn; kp += KP) {
    __syncthreads();
    if (wave == 0)
      tdm_stage_rows(ain + (size_t)(tm * 32) * Cn + kp, tA, KP, 32, Cn);
    __syncthreads();

    const bf16* ta0 = tA + ml * KP + hi * 16;
    const bf16* ta1 = ta0 + 16 * KP;
    #pragma unroll 4
    for (int k0 = 0; k0 < KP; k0 += 32) {
      __builtin_prefetch(pb0 + kp + k0 + 32, 0, 3);
      __builtin_prefetch(pb1 + kp + k0 + 32, 0, 3);
      const v16bf a0 = mk16(ld8(ta0 + k0), ld8(ta0 + k0 + 8));
      const v16bf a1 = mk16(ld8(ta1 + k0), ld8(ta1 + k0 + 8));
      const v16bf b0 = mk16(ld8(pb0 + kp + k0), ld8(pb0 + kp + k0 + 8));
      const v16bf b1 = mk16(ld8(pb1 + kp + k0), ld8(pb1 + kp + k0 + 8));
      c00 = WMMA_BF16(a0, b0, c00);
      c01 = WMMA_BF16(a0, b1, c01);
      c10 = WMMA_BF16(a1, b0, c10);
      c11 = WMMA_BF16(a1, b1, c11);
    }
  }

  const float bn0 = bp[colB0];
  const float bn1 = bp[colB1];
  #pragma unroll
  for (int r = 0; r < 8; ++r) {
    const int row0 = tm * 32 + r + 8 * hi;
    const int row1 = row0 + 16;
    out[(size_t)row0 * Cn + colB0] = c00[r] + bn0;
    out[(size_t)row0 * Cn + colB1] = c01[r] + bn1;
    out[(size_t)row1 * Cn + colB0] = c10[r] + bn0;
    out[(size_t)row1 * Cn + colB1] = c11[r] + bn1;
  }
}

// ---------------------------------------------------------------------------
extern "C" void kernel_launch(void* const* d_in, const int* in_sizes, int n_in,
                              void* d_out, int out_size, void* d_ws, size_t ws_size,
                              hipStream_t stream) {
  (void)in_sizes; (void)n_in; (void)out_size; (void)ws_size;

  const float* x  = (const float*)d_in[0];
  const float* Wq = (const float*)d_in[1];
  const float* bq = (const float*)d_in[2];
  const float* Wk = (const float*)d_in[3];
  const float* bk = (const float*)d_in[4];
  const float* Wv = (const float*)d_in[5];
  const float* bv = (const float*)d_in[6];
  const float* Wp = (const float*)d_in[7];
  const float* bp = (const float*)d_in[8];
  float* out = (float*)d_out;

  // workspace layout (bf16): xb | Wt[4] | q | k | v | att_out  (= 88 MB)
  bf16* xb = (bf16*)d_ws;
  bf16* Wt = xb + (size_t)Mrows * Cn;
  bf16* qb = Wt + (size_t)4 * Cn * Cn;
  bf16* kb = qb + (size_t)Mrows * Cn;
  bf16* vb = kb + (size_t)Mrows * Cn;
  bf16* ab = vb + (size_t)Mrows * Cn;

  xconv_kernel<<<(Mrows * Cn) / (256 * 4), 256, 0, stream>>>(x, xb);
  wconv_kernel<<<dim3(Cn / 32, Cn / 32, 4), dim3(32, 8), 0, stream>>>(Wq, Wk, Wv, Wp, Wt);

  qkv_gemm_kernel<<<dim3(1024, 3), 256, 0, stream>>>(xb, Wt, bq, bk, bv, qb, kb, vb);
  attn_kernel<<<dim3(Tn / 64, Bn * Hn), 128, 0, stream>>>(qb, kb, vb, ab);
  proj_gemm_kernel<<<1024, 256, 0, stream>>>(ab, Wt, bp, out);
}
